// SSM_28157805592813
// MI455X (gfx1250) — matrix operand order repaired and checked by reading, not compiled, not run on hardware
//
#include <hip/hip_runtime.h>

#define S_LEN   4096
#define DIM     2048
#define HID     4096
#define STATE   128
#define HEADS   64
#define HEAD_DIM 64
#define CONV    4
#define CHUNK   256
#define NCHUNK  (S_LEN / CHUNK)
#define CONV_DIM (HID + 2*STATE)        // 4352
#define D_IN    (2*HID + 2*STATE + HEADS) // 8512
#define SL      264                      // padded LDS row stride (multiple of 8)

typedef __bf16 bf16;
typedef __attribute__((ext_vector_type(16))) bf16  v16bf;
typedef __attribute__((ext_vector_type(8)))  bf16  v8bf;
typedef __attribute__((ext_vector_type(8)))  float v8f;
typedef __attribute__((ext_vector_type(4)))  unsigned int v4u;
typedef __attribute__((ext_vector_type(8)))  int v8i;
typedef __attribute__((ext_vector_type(4)))  int v4i;

#if __has_builtin(__builtin_amdgcn_tensor_load_to_lds) && __has_builtin(__builtin_amdgcn_s_wait_tensorcnt)
#define USE_TDM 1
#else
#define USE_TDM 0
#endif

static __device__ __forceinline__ bf16 f2bf(float f) {
    unsigned u = __builtin_bit_cast(unsigned, f);
    unsigned r = (u + 0x7FFFu + ((u >> 16) & 1u)) >> 16;
    unsigned short h = (unsigned short)r;
    return __builtin_bit_cast(bf16, h);
}
static __device__ __forceinline__ float bf2f(bf16 h) {
    unsigned short us = __builtin_bit_cast(unsigned short, h);
    unsigned u = ((unsigned)us) << 16;
    return __builtin_bit_cast(float, u);
}
static __device__ __forceinline__ v8f zero8() {
    v8f z;
    for (int i = 0; i < 8; i++) z[i] = 0.f;
    return z;
}
static __device__ __forceinline__ v16bf cat16(v8bf lo, v8bf hi) {
    return __builtin_shufflevector(lo, hi, 0,1,2,3,4,5,6,7,8,9,10,11,12,13,14,15);
}
static __device__ __forceinline__ v8bf ld8(const bf16* p) {
    return *reinterpret_cast<const v8bf*>(p);
}
static __device__ __forceinline__ void st8(bf16* p, v8bf v) {
    *reinterpret_cast<v8bf*>(p) = v;
}
static __device__ __forceinline__ v8f wmma_bf16(v16bf a, v16bf b, v8f c) {
    return __builtin_amdgcn_wmma_f32_16x16x32_bf16(false, a, false, b, (short)0, c, false, false);
}

#if USE_TDM
// Issue one 2-D TDM tile load: tile_rows x tile_cols bf16 elements, source row
// stride = row_stride elements, destination LDS gets 16B of pad every 64B so the
// effective LDS row stride is 40 elements (matches the fragment loaders).
static __device__ __forceinline__ void tdm_load_tile(const bf16* g, unsigned lds_addr,
                                                     unsigned tile_cols, unsigned tile_rows,
                                                     unsigned row_stride) {
    unsigned long long ga = (unsigned long long)(const void*)g;
    v4u g0;
    g0[0] = 1u;                                            // count=1 (valid user D#)
    g0[1] = lds_addr;                                      // lds_addr
    g0[2] = (unsigned)(ga & 0xFFFFFFFFu);                  // global_addr[31:0]
    g0[3] = (unsigned)((ga >> 32) & 0x1FFFFFFu) | (2u << 30); // global_addr[56:32] | type=2
    unsigned dim0 = row_stride;                            // tensor_dim0
    unsigned dim1 = 0x100000u;                             // tensor_dim1 (ample rows)
    unsigned long long st0 = (unsigned long long)row_stride; // tensor_dim0_stride
    v8i g1;
    g1[0] = (int)((1u << 16) |     // data_size = 2 bytes
                  (1u << 20) |     // pad_enable
                  (3u << 22) |     // pad_interval: every 16 DWORDs (64B)
                  (3u << 25));     // pad_amount: 4 DWORDs (16B)
    g1[1] = (int)((dim0 & 0xFFFFu) << 16);                             // dim0[15:0]
    g1[2] = (int)(((dim0 >> 16) & 0xFFFFu) | ((dim1 & 0xFFFFu) << 16));
    g1[3] = (int)(((dim1 >> 16) & 0xFFFFu) | ((tile_cols & 0xFFFFu) << 16)); // tile_dim0
    g1[4] = (int)(tile_rows & 0xFFFFu);                                // tile_dim1, tile_dim2=0
    g1[5] = (int)(unsigned)(st0 & 0xFFFFFFFFu);
    g1[6] = (int)(unsigned)((st0 >> 32) & 0xFFFFu);                    // stride0 hi, stride1 lo=0
    g1[7] = 0;
    v4i g2 = {0, 0, 0, 0};
    v4i g3 = {0, 0, 0, 0};
#if defined(__clang_major__) && (__clang_major__ >= 23)
    v8i g4 = {0, 0, 0, 0, 0, 0, 0, 0};
    __builtin_amdgcn_tensor_load_to_lds(g0, g1, g2, g3, g4, 0);
#else
    __builtin_amdgcn_tensor_load_to_lds(g0, g1, g2, g3, 0);
#endif
}
static __device__ __forceinline__ unsigned lds_off(const void* p) {
    return (unsigned)(unsigned long long)(uintptr_t)p;
}
#endif

// ---------------------------------------------------------------------------
// f32 -> bf16 convert
// ---------------------------------------------------------------------------
__global__ __launch_bounds__(256) void cvt_f32_bf16(const float* __restrict__ src,
                                                    bf16* __restrict__ dst, int n) {
    int i = blockIdx.x * 256 + threadIdx.x;
    if (i < n) dst[i] = f2bf(src[i]);
}

// ---------------------------------------------------------------------------
// Generic GEMM: C[M,N] (f32) = A[M,K] (bf16, row-major) x B[N,K]^T (bf16 row-major)
// Block tile 256(M) x 64(N), K-step 32, 8 waves each owning a 32x64 strip
// (2x4 wmma accumulators -> 8 wmma per K-step per wave).
// Tiles staged to LDS by the Tensor Data Mover (double buffered, TENSORcnt
// pipelined); synchronous global->LDS fallback when the builtin is absent.
// ---------------------------------------------------------------------------
__global__ __launch_bounds__(256) void gemm_bf16_nt(const bf16* __restrict__ A,
                                                    const bf16* __restrict__ B,
                                                    float* __restrict__ C,
                                                    int M, int N, int K) {
    __shared__ __attribute__((aligned(16))) bf16 As[2][256 * 40];
    __shared__ __attribute__((aligned(16))) bf16 Bs[2][64 * 40];
    const int tid  = threadIdx.x;
    const int lane = tid & 31, wave = tid >> 5;
    const int m0 = blockIdx.y * 256, n0 = blockIdx.x * 64;
    const int kh = lane >> 4, ln15 = lane & 15;
    (void)M;

    v8f acc[2][4];
    for (int i = 0; i < 2; i++)
        for (int j = 0; j < 4; j++) acc[i][j] = zero8();

    const int nb = K / 32;

#if USE_TDM
    if (wave == 0) {
        tdm_load_tile(A + (size_t)m0 * K, lds_off(&As[0][0]), 32, 256, (unsigned)K);
        tdm_load_tile(B + (size_t)n0 * K, lds_off(&Bs[0][0]), 32, 64, (unsigned)K);
    }
#else
    {   // synchronous stage of k-block 0 into buffer 0
        for (int i = 0; i < 4; i++) {
            int idx = tid + (i << 8);
            int r = idx >> 2, c8 = (idx & 3) << 3;
            st8(&As[0][r * 40 + c8], ld8(A + (size_t)(m0 + r) * K + c8));
        }
        int r = tid >> 2, c8 = (tid & 3) << 3;
        st8(&Bs[0][r * 40 + c8], ld8(B + (size_t)(n0 + r) * K + c8));
    }
#endif

    for (int kb = 0; kb < nb; kb++) {
        const int cur = kb & 1, nxt = cur ^ 1;
        if (kb + 1 < nb) {
            int k1 = (kb + 1) * 32;
#if USE_TDM
            if (wave == 0) {
                tdm_load_tile(A + (size_t)m0 * K + k1, lds_off(&As[nxt][0]), 32, 256, (unsigned)K);
                tdm_load_tile(B + (size_t)n0 * K + k1, lds_off(&Bs[nxt][0]), 32, 64, (unsigned)K);
            }
#else
            for (int i = 0; i < 4; i++) {
                int idx = tid + (i << 8);
                int r = idx >> 2, c8 = (idx & 3) << 3;
                st8(&As[nxt][r * 40 + c8], ld8(A + (size_t)(m0 + r) * K + k1 + c8));
            }
            int r = tid >> 2, c8 = (tid & 3) << 3;
            st8(&Bs[nxt][r * 40 + c8], ld8(B + (size_t)(n0 + r) * K + k1 + c8));
#endif
        }
#if USE_TDM
        // TENSORcnt decrements in order: <=2 outstanding means the current
        // stage's two tile loads have landed in LDS. (Immediate-only builtin.)
        if (kb + 1 < nb) __builtin_amdgcn_s_wait_tensorcnt(2);
        else             __builtin_amdgcn_s_wait_tensorcnt(0);
#endif
        __syncthreads();

        v16bf af[2], bfr[4];
#pragma unroll
        for (int tm = 0; tm < 2; tm++) {
            const bf16* p = &As[cur][(wave * 32 + tm * 16 + ln15) * 40 + kh * 8];
            af[tm] = cat16(ld8(p), ld8(p + 16));
        }
#pragma unroll
        for (int tn = 0; tn < 4; tn++) {
            const bf16* p = &Bs[cur][(tn * 16 + ln15) * 40 + kh * 8];
            bfr[tn] = cat16(ld8(p), ld8(p + 16));
        }
#pragma unroll
        for (int tm = 0; tm < 2; tm++)
#pragma unroll
            for (int tn = 0; tn < 4; tn++)
                acc[tm][tn] = wmma_bf16(af[tm], bfr[tn], acc[tm][tn]);
        __syncthreads();
    }
#pragma unroll
    for (int tm = 0; tm < 2; tm++)
#pragma unroll
        for (int tn = 0; tn < 4; tn++)
#pragma unroll
            for (int j = 0; j < 8; j++) {
                int r  = m0 + wave * 32 + tm * 16 + j + 8 * kh;
                int cc = n0 + tn * 16 + ln15;
                C[(size_t)r * N + cc] = acc[tm][tn][j];
            }
}

// ---------------------------------------------------------------------------
// causal conv1d + SiLU, split into xh / B / C (bf16)
// ---------------------------------------------------------------------------
__global__ __launch_bounds__(256) void conv_silu_kernel(const float* __restrict__ zxbcdt,
                                                        const float* __restrict__ convw,
                                                        bf16* __restrict__ xh,
                                                        bf16* __restrict__ Bb,
                                                        bf16* __restrict__ Cb) {
    int idx = blockIdx.x * 256 + threadIdx.x;
    if (idx >= S_LEN * CONV_DIM) return;
    int t = idx / CONV_DIM, ci = idx - t * CONV_DIM;
    float acc = 0.f;
#pragma unroll
    for (int k = 0; k < CONV; k++) {
        int ts = t - (CONV - 1) + k;
        if (ts >= 0) acc += zxbcdt[(size_t)ts * D_IN + HID + ci] * convw[ci * CONV + k];
    }
    float v = acc / (1.f + expf(-acc));  // silu
    if (ci < HID)               xh[(size_t)t * HID + ci] = f2bf(v);
    else if (ci < HID + STATE)  Bb[t * STATE + (ci - HID)] = f2bf(v);
    else                        Cb[t * STATE + (ci - HID - STATE)] = f2bf(v);
}

// ---------------------------------------------------------------------------
// dt = softplus(dt_raw + bias); dA = -exp(A_log) * dt
// ---------------------------------------------------------------------------
__global__ __launch_bounds__(256) void dt_kernel(const float* __restrict__ zxbcdt,
                                                 const float* __restrict__ dt_bias,
                                                 const float* __restrict__ A_log,
                                                 float* __restrict__ dts,
                                                 float* __restrict__ dAv) {
    int idx = blockIdx.x * 256 + threadIdx.x;
    if (idx >= S_LEN * HEADS) return;
    int t = idx >> 6, h = idx & 63;
    float x = zxbcdt[(size_t)t * D_IN + HID + CONV_DIM + h] + dt_bias[h];
    float sp = (x > 20.f) ? x : log1pf(expf(x));
    dts[idx] = sp;
    dAv[idx] = -expf(A_log[h]) * sp;
}

// ---------------------------------------------------------------------------
// per-(chunk, head) inclusive cumsum of dA over the 256-long chunk
// ---------------------------------------------------------------------------
__global__ __launch_bounds__(256) void cumsum_kernel(const float* __restrict__ dAv,
                                                     float* __restrict__ dAc) {
    __shared__ float buf[CHUNK];
    int c = blockIdx.x >> 6, h = blockIdx.x & 63;
    int l = threadIdx.x;
    int t = c * CHUNK + l;
    buf[l] = dAv[t * HEADS + h];
    __syncthreads();
    for (int off = 1; off < CHUNK; off <<= 1) {
        float add = (l >= off) ? buf[l - off] : 0.f;
        __syncthreads();
        buf[l] += add;
        __syncthreads();
    }
    dAc[t * HEADS + h] = buf[l];
}

// ---------------------------------------------------------------------------
// chunk states: states[c,h,p,n] = sum_l (x*dt*exp(dA_last - dAc))[l,p] * B[l,n]
// one block per (c,h); WMMA over K = l (256)
// ---------------------------------------------------------------------------
__global__ __launch_bounds__(256) void states_kernel(const bf16* __restrict__ xh,
                                                     const bf16* __restrict__ Bb,
                                                     const float* __restrict__ dts,
                                                     const float* __restrict__ dAc,
                                                     float* __restrict__ states) {
    __shared__ __attribute__((aligned(16))) bf16 Xs[HEAD_DIM * SL]; // [p][l]
    __shared__ __attribute__((aligned(16))) bf16 Bs[STATE * SL];    // [n][l]
    __shared__ float scl[CHUNK];
    int c = blockIdx.x >> 6, h = blockIdx.x & 63;
    int tid = threadIdx.x, lane = tid & 31, wave = tid >> 5;
    int tbase = c * CHUNK;
    float dAlast = dAc[(tbase + CHUNK - 1) * HEADS + h];
    if (tid < CHUNK) {
        int t = tbase + tid;
        scl[tid] = dts[t * HEADS + h] * expf(dAlast - dAc[t * HEADS + h]);
    }
    __syncthreads();
    for (int i = 0; i < 64; i++) {                  // Xs[p][l] = xh * scl
        int p = tid & 63, l = i * 4 + (tid >> 6);
        int t = tbase + l;
        Xs[p * SL + l] = f2bf(bf2f(xh[(size_t)t * HID + h * HEAD_DIM + p]) * scl[l]);
    }
    for (int i = 0; i < 128; i++) {                 // Bs[n][l] (transpose)
        int n = tid & 127, l = i * 2 + (tid >> 7);
        int t = tbase + l;
        Bs[n * SL + l] = Bb[t * STATE + n];
    }
    __syncthreads();

    const int pt = wave >> 1, nh = wave & 1;
    const int kh = lane >> 4, ln15 = lane & 15;
    v8f acc[4];
    for (int i = 0; i < 4; i++) acc[i] = zero8();
    for (int ks = 0; ks < CHUNK / 32; ks++) {
        const bf16* pa = &Xs[(pt * 16 + ln15) * SL + ks * 32 + kh * 8];
        v16bf af = cat16(ld8(pa), ld8(pa + 16));
#pragma unroll
        for (int i = 0; i < 4; i++) {
            int n = (nh * 4 + i) * 16 + ln15;
            const bf16* pb = &Bs[n * SL + ks * 32 + kh * 8];
            v16bf bfr = cat16(ld8(pb), ld8(pb + 16));
            acc[i] = wmma_bf16(af, bfr, acc[i]);
        }
    }
    size_t base = (size_t)(c * HEADS + h) * HEAD_DIM * STATE;
#pragma unroll
    for (int i = 0; i < 4; i++)
#pragma unroll
        for (int j = 0; j < 8; j++) {
            int p = pt * 16 + j + 8 * kh;
            int n = (nh * 4 + i) * 16 + ln15;
            states[base + (size_t)p * STATE + n] = acc[i][j];
        }
}

// ---------------------------------------------------------------------------
// sequential inter-chunk scan: prev[c] = carry; carry = carry*decay + states[c]
// emits prev in bf16 for WMMA consumption downstream
// ---------------------------------------------------------------------------
__global__ __launch_bounds__(256) void chunkscan_kernel(const float* __restrict__ states,
                                                        const float* __restrict__ dAc,
                                                        bf16* __restrict__ prevbf) {
    int h = blockIdx.x;
    int base_pn = threadIdx.x * 32;
    float carry[32];
#pragma unroll
    for (int k = 0; k < 32; k++) carry[k] = 0.f;
    for (int c = 0; c < NCHUNK; c++) {
        float dec = expf(dAc[(c * CHUNK + CHUNK - 1) * HEADS + h]);
        size_t off = (size_t)(c * HEADS + h) * HEAD_DIM * STATE + base_pn;
#pragma unroll
        for (int k = 0; k < 32; k++) {
            float st = states[off + k];
            prevbf[off + k] = f2bf(carry[k]);
            carry[k] = carry[k] * dec + st;
        }
    }
}

// ---------------------------------------------------------------------------
// y = y_diag + y_off + x*D per (chunk, head). 8 waves each own a 32-row l-strip.
//   y_off  = exp(dAc[l]) * (C @ prev^T)
//   y_diag = mask(exp(dAc[l]-dAc[s]) * (C @ B^T)) @ (x*dt)
// G fragments are remapped D->A through per-wave LDS scratch.
// ---------------------------------------------------------------------------
__global__ __launch_bounds__(256) void ssd_y_kernel(const bf16* __restrict__ xh,
                                                    const bf16* __restrict__ Bb,
                                                    const bf16* __restrict__ Cb,
                                                    const float* __restrict__ dts,
                                                    const float* __restrict__ dAc,
                                                    const bf16* __restrict__ prevbf,
                                                    const float* __restrict__ Dp,
                                                    float* __restrict__ yf) {
    __shared__ __attribute__((aligned(16))) bf16 Xs[HEAD_DIM * SL];  // [p][l], x*dt
    __shared__ __attribute__((aligned(16))) bf16 Mb[8 * 32 * 32];    // per-wave M scratch
    __shared__ float dAcs[CHUNK];
    int c = blockIdx.x >> 6, h = blockIdx.x & 63;
    int tid = threadIdx.x, lane = tid & 31, w = tid >> 5;
    int tbase = c * CHUNK;
    if (tid < CHUNK) dAcs[tid] = dAc[(tbase + tid) * HEADS + h];
    for (int i = 0; i < 64; i++) {
        int p = tid & 63, l = i * 4 + (tid >> 6);
        int t = tbase + l;
        Xs[p * SL + l] = f2bf(bf2f(xh[(size_t)t * HID + h * HEAD_DIM + p]) * dts[t * HEADS + h]);
    }
    __syncthreads();

    const int kh = lane >> 4, ln15 = lane & 15;
    v8f acc[2][4];
    for (int i = 0; i < 2; i++)
        for (int j = 0; j < 4; j++) acc[i][j] = zero8();

    // ---- y_off = C @ prev^T ----
    for (int ks = 0; ks < STATE / 32; ks++) {
        v16bf af[2];
#pragma unroll
        for (int lt = 0; lt < 2; lt++) {
            int t = tbase + w * 32 + lt * 16 + ln15;
            const bf16* p = &Cb[(size_t)t * STATE + ks * 32 + kh * 8];
            af[lt] = cat16(ld8(p), ld8(p + 16));
        }
#pragma unroll
        for (int pt = 0; pt < 4; pt++) {
            const bf16* pb = &prevbf[((size_t)(c * HEADS + h) * HEAD_DIM + pt * 16 + ln15) * STATE
                                     + ks * 32 + kh * 8];
            v16bf bfr = cat16(ld8(pb), ld8(pb + 16));
#pragma unroll
            for (int lt = 0; lt < 2; lt++)
                acc[lt][pt] = wmma_bf16(af[lt], bfr, acc[lt][pt]);
        }
    }
    // scale y_off by exp(dAc[l])
    float el[2][8];
#pragma unroll
    for (int lt = 0; lt < 2; lt++)
#pragma unroll
        for (int j = 0; j < 8; j++)
            el[lt][j] = expf(dAcs[w * 32 + lt * 16 + j + 8 * kh]);
#pragma unroll
    for (int lt = 0; lt < 2; lt++)
#pragma unroll
        for (int pt = 0; pt < 4; pt++)
#pragma unroll
            for (int j = 0; j < 8; j++)
                acc[lt][pt][j] *= el[lt][j];

    // ---- y_diag: loop over causal s-blocks of 32 ----
    for (int sb = 0; sb <= w; sb++) {
        v8f g[2][2];
        for (int i = 0; i < 2; i++)
            for (int j = 0; j < 2; j++) g[i][j] = zero8();
        for (int ks = 0; ks < STATE / 32; ks++) {
            v16bf af[2], bb[2];
#pragma unroll
            for (int lt = 0; lt < 2; lt++) {
                int t = tbase + w * 32 + lt * 16 + ln15;
                const bf16* p = &Cb[(size_t)t * STATE + ks * 32 + kh * 8];
                af[lt] = cat16(ld8(p), ld8(p + 16));
            }
#pragma unroll
            for (int st = 0; st < 2; st++) {
                int ts = tbase + sb * 32 + st * 16 + ln15;
                const bf16* p = &Bb[(size_t)ts * STATE + ks * 32 + kh * 8];
                bb[st] = cat16(ld8(p), ld8(p + 16));
            }
#pragma unroll
            for (int lt = 0; lt < 2; lt++)
#pragma unroll
                for (int st = 0; st < 2; st++)
                    g[lt][st] = wmma_bf16(af[lt], bb[st], g[lt][st]);
        }
        // causal mask + decay, store as bf16 into per-wave scratch (D->A remap)
#pragma unroll
        for (int lt = 0; lt < 2; lt++)
#pragma unroll
            for (int st = 0; st < 2; st++)
#pragma unroll
                for (int j = 0; j < 8; j++) {
                    int lr = w * 32 + lt * 16 + j + 8 * kh;
                    int sc = sb * 32 + st * 16 + ln15;
                    float v = (sc <= lr) ? g[lt][st][j] * expf(dAcs[lr] - dAcs[sc]) : 0.f;
                    Mb[(w * 32 + lt * 16 + j + 8 * kh) * 32 + st * 16 + ln15] = f2bf(v);
                }
        // LDS ops are in-order within a wave; fence the compiler + DScnt anyway
        asm volatile("s_wait_dscnt 0x0" ::: "memory");
        v16bf am[2];
#pragma unroll
        for (int lt = 0; lt < 2; lt++) {
            const bf16* p = &Mb[(w * 32 + lt * 16 + ln15) * 32 + kh * 8];
            am[lt] = cat16(ld8(p), ld8(p + 16));
        }
#pragma unroll
        for (int pt = 0; pt < 4; pt++) {
            const bf16* px = &Xs[(pt * 16 + ln15) * SL + sb * 32 + kh * 8];
            v16bf bx = cat16(ld8(px), ld8(px + 16));
#pragma unroll
            for (int lt = 0; lt < 2; lt++)
                acc[lt][pt] = wmma_bf16(am[lt], bx, acc[lt][pt]);
        }
    }

    // ---- store y + x*D ----
    float Dh = Dp[h];
#pragma unroll
    for (int lt = 0; lt < 2; lt++)
#pragma unroll
        for (int pt = 0; pt < 4; pt++)
#pragma unroll
            for (int j = 0; j < 8; j++) {
                int l = w * 32 + lt * 16 + j + 8 * kh;
                int t = tbase + l;
                int ch = h * HEAD_DIM + pt * 16 + ln15;
                yf[(size_t)t * HID + ch] = acc[lt][pt][j] + bf2f(xh[(size_t)t * HID + ch]) * Dh;
            }
}

// ---------------------------------------------------------------------------
// y = y * silu(z); RMSNorm; -> bf16 for final GEMM
// ---------------------------------------------------------------------------
__global__ __launch_bounds__(256) void gated_rmsnorm_kernel(const float* __restrict__ yf,
                                                            const float* __restrict__ zxbcdt,
                                                            const float* __restrict__ norm_w,
                                                            bf16* __restrict__ ybf) {
    __shared__ float red[256];
    int t = blockIdx.x, tid = threadIdx.x;
    float y2[16];
    float ss = 0.f;
#pragma unroll
    for (int i = 0; i < 16; i++) {
        int ch = tid + i * 256;
        float z = zxbcdt[(size_t)t * D_IN + ch];
        float v = yf[(size_t)t * HID + ch] * (z / (1.f + expf(-z)));
        y2[i] = v;
        ss += v * v;
    }
    red[tid] = ss;
    __syncthreads();
    for (int off = 128; off > 0; off >>= 1) {
        if (tid < off) red[tid] += red[tid + off];
        __syncthreads();
    }
    float rs = rsqrtf(red[0] / (float)HID + 1e-5f);
#pragma unroll
    for (int i = 0; i < 16; i++) {
        int ch = tid + i * 256;
        ybf[(size_t)t * HID + ch] = f2bf(y2[i] * rs * norm_w[ch]);
    }
}

// ---------------------------------------------------------------------------
extern "C" void kernel_launch(void* const* d_in, const int* in_sizes, int n_in,
                              void* d_out, int out_size, void* d_ws, size_t ws_size,
                              hipStream_t stream) {
    (void)in_sizes; (void)n_in; (void)out_size; (void)ws_size;
    const float* x       = (const float*)d_in[0];
    const float* in_w    = (const float*)d_in[1];
    const float* conv_w  = (const float*)d_in[2];
    const float* dt_bias = (const float*)d_in[3];
    const float* A_log   = (const float*)d_in[4];
    const float* Dp      = (const float*)d_in[5];
    const float* norm_w  = (const float*)d_in[6];
    const float* out_w   = (const float*)d_in[7];
    float* out = (float*)d_out;

    char* ws = (char*)d_ws;
    size_t off = 0;
    auto alloc = [&](size_t bytes) -> char* {
        char* p = ws + off;
        off += (bytes + 255) & ~(size_t)255;
        return p;
    };
    bf16*  xb     = (bf16*) alloc((size_t)S_LEN * DIM * 2);
    bf16*  win    = (bf16*) alloc((size_t)D_IN * DIM * 2);
    bf16*  wout   = (bf16*) alloc((size_t)DIM * HID * 2);
    float* zxbcdt = (float*)alloc((size_t)S_LEN * D_IN * 4);
    bf16*  xh     = (bf16*) alloc((size_t)S_LEN * HID * 2);
    bf16*  Bb     = (bf16*) alloc((size_t)S_LEN * STATE * 2);
    bf16*  Cb     = (bf16*) alloc((size_t)S_LEN * STATE * 2);
    float* dts    = (float*)alloc((size_t)S_LEN * HEADS * 4);
    float* dAv    = (float*)alloc((size_t)S_LEN * HEADS * 4);
    float* dAc    = (float*)alloc((size_t)S_LEN * HEADS * 4);
    float* states = (float*)alloc((size_t)NCHUNK * HEADS * HEAD_DIM * STATE * 4);
    bf16*  prevbf = (bf16*) alloc((size_t)NCHUNK * HEADS * HEAD_DIM * STATE * 2);
    float* yf     = (float*)alloc((size_t)S_LEN * HID * 4);
    bf16*  ybf    = (bf16*) alloc((size_t)S_LEN * HID * 2);

    cvt_f32_bf16<<<(S_LEN * DIM + 255) / 256, 256, 0, stream>>>(x, xb, S_LEN * DIM);
    cvt_f32_bf16<<<(D_IN * DIM + 255) / 256, 256, 0, stream>>>(in_w, win, D_IN * DIM);
    cvt_f32_bf16<<<(DIM * HID + 255) / 256, 256, 0, stream>>>(out_w, wout, DIM * HID);

    gemm_bf16_nt<<<dim3(D_IN / 64, S_LEN / 256), 256, 0, stream>>>(xb, win, zxbcdt,
                                                                   S_LEN, D_IN, DIM);
    conv_silu_kernel<<<(S_LEN * CONV_DIM + 255) / 256, 256, 0, stream>>>(zxbcdt, conv_w,
                                                                         xh, Bb, Cb);
    dt_kernel<<<(S_LEN * HEADS + 255) / 256, 256, 0, stream>>>(zxbcdt, dt_bias, A_log,
                                                               dts, dAv);
    cumsum_kernel<<<NCHUNK * HEADS, 256, 0, stream>>>(dAv, dAc);
    states_kernel<<<NCHUNK * HEADS, 256, 0, stream>>>(xh, Bb, dts, dAc, states);
    chunkscan_kernel<<<HEADS, 256, 0, stream>>>(states, dAc, prevbf);
    ssd_y_kernel<<<NCHUNK * HEADS, 256, 0, stream>>>(xh, Bb, Cb, dts, dAc, prevbf, Dp, yf);
    gated_rmsnorm_kernel<<<S_LEN, 256, 0, stream>>>(yf, zxbcdt, norm_w, ybf);
    gemm_bf16_nt<<<dim3(DIM / 64, S_LEN / 256), 256, 0, stream>>>(ybf, wout, out,
                                                                  S_LEN, DIM, HID);
}
